// LSTM_3882650436141
// MI455X (gfx1250) — matrix operand order repaired and checked by reading, not compiled, not run on hardware
//
#include <hip/hip_runtime.h>
#include <hip/hip_bf16.h>

// ---------------- problem constants ----------------
#define T_   512
#define B_   128
#define IN_  800      // == 2*H_, every layer input is 800 wide
#define H_   400
#define L_   3
#define G4   1600     // 4*H
#define KP   416      // H padded to multiple of 32 for WMMA K
#define MROWS (T_ * B_)   // 65536
#define NBLK 25           // blocks per direction in scan kernel (one per 16 hidden cols)

typedef __attribute__((ext_vector_type(16))) __bf16 bf16x16;
typedef __attribute__((ext_vector_type(8)))  float  f32x8;
typedef unsigned int u32x4_t __attribute__((ext_vector_type(4)));
typedef int          i32x4_t __attribute__((ext_vector_type(4)));
typedef int          i32x8_t __attribute__((ext_vector_type(8)));

#if defined(__has_builtin)
#if __has_builtin(__builtin_amdgcn_tensor_load_to_lds) && __has_builtin(__builtin_amdgcn_s_wait_tensorcnt)
#define HAVE_TDM 1
#endif
#endif

union BFrag { bf16x16 v; uint4 u[2]; };

// ---------------- WMMA helpers (layouts per CDNA5 ISA 7.12.2) ----------------
__device__ __forceinline__ f32x8 wmma_bf16(bf16x16 a, bf16x16 b, f32x8 c) {
  return __builtin_amdgcn_wmma_f32_16x16x32_bf16(
      /*neg_a=*/false, a, /*neg_b=*/false, b,
      /*c_mod=*/(short)0, c, /*reuse_a=*/false, /*reuse_b=*/false);
}

// A: 16x32 tile (MxK) from row-major [rows, lda] bf16.
__device__ __forceinline__ bf16x16 load_a_frag(const __hip_bfloat16* __restrict__ A,
                                               int lda, int row0, int k0, int lane) {
  const __hip_bfloat16* p =
      A + (size_t)(row0 + (lane & 15)) * lda + k0 + ((lane >> 4) * 8);
  BFrag r;
  r.u[0] = *reinterpret_cast<const uint4*>(p);
  r.u[1] = *reinterpret_cast<const uint4*>(p + 16);
  return r.v;
}

// B: 32x16 tile (KxN) where B[k][n] = W[n][k], W row-major [N, ldw] bf16.
__device__ __forceinline__ bf16x16 load_b_frag(const __hip_bfloat16* __restrict__ W,
                                               int ldw, int col0, int k0, int lane) {
  const __hip_bfloat16* p =
      W + (size_t)(col0 + (lane & 15)) * ldw + k0 + ((lane >> 4) * 8);
  BFrag r;
  r.u[0] = *reinterpret_cast<const uint4*>(p);
  r.u[1] = *reinterpret_cast<const uint4*>(p + 16);
  return r.v;
}

// C/D tile: element r of lane -> (M = row0 + r + 8*(lane>=16), N = col0 + (lane&15))
__device__ __forceinline__ void store_c_tile(float* __restrict__ out, int ldc,
                                             int row0, int col0, int lane, f32x8 c) {
  const int n  = col0 + (lane & 15);
  const int rb = row0 + ((lane >> 4) * 8);
  float* p = out + (size_t)rb * ldc + n;
#pragma unroll
  for (int r = 0; r < 8; ++r) p[(size_t)r * ldc] = c[r];
}

__device__ __forceinline__ f32x8 load_c_tile(const float* __restrict__ in, int ldc,
                                             int row0, int col0, int lane) {
  const int n  = col0 + (lane & 15);
  const int rb = row0 + ((lane >> 4) * 8);
  const float* p = in + (size_t)rb * ldc + n;
  f32x8 c;
#pragma unroll
  for (int r = 0; r < 8; ++r) c[r] = p[(size_t)r * ldc];
  return c;
}

// ---------------- conversion / setup kernels ----------------
__global__ void cvt_f32_bf16_kernel(const float* __restrict__ src,
                                    __hip_bfloat16* __restrict__ dst, long long n) {
  long long i = (long long)blockIdx.x * blockDim.x + threadIdx.x;
  long long stride = (long long)gridDim.x * blockDim.x;
  for (; i < n; i += stride) dst[i] = __float2bfloat16(src[i]);
}

__global__ void cvt_pad_whh_kernel(const float* __restrict__ src,
                                   __hip_bfloat16* __restrict__ dst, int rows) {
  long long n = (long long)rows * KP;
  long long i = (long long)blockIdx.x * blockDim.x + threadIdx.x;
  long long stride = (long long)gridDim.x * blockDim.x;
  for (; i < n; i += stride) {
    int r = (int)(i / KP), cc = (int)(i % KP);
    dst[i] = (cc < H_) ? __float2bfloat16(src[(size_t)r * H_ + cc])
                       : __float2bfloat16(0.f);
  }
}

__global__ void make_bias_kernel(const float* __restrict__ bih,
                                 const float* __restrict__ bhh,
                                 float* __restrict__ b, int n) {
  int i = blockIdx.x * blockDim.x + threadIdx.x;
  if (i < n) b[i] = bih[i] + bhh[i];
}

__global__ void zero_ctrs_kernel(unsigned* __restrict__ c, int n) {
  int i = blockIdx.x * blockDim.x + threadIdx.x;
  if (i < n) c[i] = 0u;
}

// ---------------- xproj GEMM: gates[dir] = X @ Wih[dir]^T + bias ----------------
// block = 256 thr (8 waves), tile 256(M) x 64(N), wave tile 32x64 (2x4 WMMA tiles)
__global__ __launch_bounds__(256) void xproj_gemm_kernel(
    const __hip_bfloat16* __restrict__ X,
    const __hip_bfloat16* __restrict__ Wih,
    const float* __restrict__ bias,
    float* __restrict__ gates) {
  const int dir  = blockIdx.z;
  const __hip_bfloat16* W = Wih + (size_t)dir * G4 * IN_;
  const float* bs = bias + (size_t)dir * G4;
  float* Gd = gates + (size_t)dir * MROWS * G4;

  const int lane = threadIdx.x & 31;
  const int wave = threadIdx.x >> 5;
  const int row0 = blockIdx.x * 256 + wave * 32;  // 8 wave rows of 32
  const int col0 = blockIdx.y * 64;               // all waves share 64 cols

  f32x8 c[2][4];
#pragma unroll
  for (int ni = 0; ni < 4; ++ni) {
    float bv = bs[col0 + ni * 16 + (lane & 15)];
#pragma unroll
    for (int mi = 0; mi < 2; ++mi)
#pragma unroll
      for (int r = 0; r < 8; ++r) c[mi][ni][r] = bv;
  }

  for (int k = 0; k < IN_; k += 32) {
    if (k + 32 < IN_) {  // hint next K-slab into cache (global_prefetch_b8)
      __builtin_prefetch(X + (size_t)(row0 + (lane & 15)) * IN_ + k + 32, 0, 1);
      __builtin_prefetch(W + (size_t)(col0 + (lane & 15)) * IN_ + k + 32, 0, 1);
    }
    bf16x16 a0 = load_a_frag(X, IN_, row0,      k, lane);
    bf16x16 a1 = load_a_frag(X, IN_, row0 + 16, k, lane);
#pragma unroll
    for (int ni = 0; ni < 4; ++ni) {
      bf16x16 b = load_b_frag(W, IN_, col0 + ni * 16, k, lane);
      c[0][ni] = wmma_bf16(a0, b, c[0][ni]);
      c[1][ni] = wmma_bf16(a1, b, c[1][ni]);
    }
  }

#pragma unroll
  for (int mi = 0; mi < 2; ++mi)
#pragma unroll
    for (int ni = 0; ni < 4; ++ni)
      store_c_tile(Gd, G4, row0 + mi * 16, col0 + ni * 16, lane, c[mi][ni]);
}

// ---------------- persistent recurrent scan ----------------
__device__ __forceinline__ void half_grid_barrier(unsigned* ctr, unsigned nblk,
                                                  unsigned phase) {
  __syncthreads();
  if (threadIdx.x == 0) {
    __threadfence();
    __hip_atomic_fetch_add(ctr, 1u, __ATOMIC_RELEASE, __HIP_MEMORY_SCOPE_AGENT);
    const unsigned target = nblk * phase;
    while (__hip_atomic_load(ctr, __ATOMIC_ACQUIRE, __HIP_MEMORY_SCOPE_AGENT) < target)
      __builtin_amdgcn_s_sleep(1);
  }
  __syncthreads();
}

__device__ __forceinline__ float fsig(float x) { return 1.f / (1.f + __expf(-x)); }

// grid = 50 blocks: [0,25) forward, [25,50) backward. Block owns 16 hidden
// columns: its 8 waves each compute ALL FOUR gate tiles (i,f,g,o) for a
// 16-row batch tile, then do the LSTM pointwise entirely in registers.
// h state ping-pongs between two buffers so ONE grid barrier per step suffices.
__global__ __launch_bounds__(256) void lstm_scan_kernel(
    const float* __restrict__ gates,          // [2][MROWS][G4]  xproj + bias
    const __hip_bfloat16* __restrict__ Whh,   // [2][G4][KP] bf16, K-padded
    __hip_bfloat16* __restrict__ hping,       // [2 dirs][2 parity][B_][KP]
    float* __restrict__ cst,                  // [2][B_][H_]
    __hip_bfloat16* __restrict__ Xout,        // [T_][B_][800] bf16 or null
    float* __restrict__ Fout,                 // [T_][B_][800] f32 or null
    const int* __restrict__ bsz,              // [T_]
    unsigned* __restrict__ ctrs) {            // [2]
  const int blk  = blockIdx.x;
  const int dir  = blk / NBLK;
  const int nt   = blk % NBLK;
  const int tid  = threadIdx.x;
  const int lane = tid & 31;
  const int wave = tid >> 5;

  const float* Gd = gates + (size_t)dir * MROWS * G4;
  const __hip_bfloat16* Wd = Whh + (size_t)dir * G4 * KP;
  __hip_bfloat16* h0 = hping + (size_t)dir * 2 * B_ * KP;
  __hip_bfloat16* h1 = h0 + (size_t)B_ * KP;
  float* cd = cst + (size_t)dir * B_ * H_;
  unsigned* ctr = ctrs + dir;

  // 4 gate groups x 16 rows x KP cols of Whh, resident for all 512 steps
  __shared__ __align__(16) __hip_bfloat16 sW[64 * KP];  // 52 KB of 320 KB LDS

  const int ht0 = nt * 16;  // hidden columns [ht0, ht0+16)

#if HAVE_TDM
  if (wave == 0) {
    // Four 16x416 2D tiles via the Tensor Data Mover (D# per ISA 8.3/8.4).
#pragma unroll
    for (int g = 0; g < 4; ++g) {
      const unsigned lds_addr =
          (unsigned)(unsigned long long)(uintptr_t)&sW[(size_t)g * 16 * KP];
      const unsigned long long ga =
          (unsigned long long)(uintptr_t)(Wd + (size_t)(g * H_ + ht0) * KP);
      u32x4_t d0;
      d0[0] = 1u;                                   // count=1, user descriptor
      d0[1] = lds_addr;                             // lds_addr[31:0]
      d0[2] = (unsigned)(ga & 0xffffffffu);         // global_addr[31:0]
      d0[3] = (unsigned)((ga >> 32) & 0x01ffffffu)  // global_addr[56:32]
              | (2u << 30);                         // type=2 ("image")
      i32x8_t d1;
      d1[0] = (int)(1u << 16);                      // data_size=1 (2 bytes)
      d1[1] = (int)((unsigned)KP << 16);            // tensor_dim0[15:0]=416
      d1[2] = (int)(16u << 16);                     // dim0 hi=0 | tensor_dim1=16
      d1[3] = (int)((unsigned)KP << 16);            // dim1 hi=0 | tile_dim0=416
      d1[4] = 16;                                   // tile_dim1=16, tile_dim2=0
      d1[5] = KP;                                   // tensor_dim0_stride=416
      d1[6] = 0; d1[7] = 0;
      i32x4_t z4 = {0, 0, 0, 0};
#if __clang_major__ >= 23
      i32x8_t z8 = {0, 0, 0, 0, 0, 0, 0, 0};
      __builtin_amdgcn_tensor_load_to_lds(d0, d1, z4, z4, z8, 0);
#else
      __builtin_amdgcn_tensor_load_to_lds(d0, d1, z4, z4, 0);
#endif
    }
    __builtin_amdgcn_s_wait_tensorcnt(0);
  }
#else
  for (int i = tid; i < 64 * KP; i += 256) {
    const int r = i / KP, cc = i % KP;
    const int gcol = (r >> 4) * H_ + ht0 + (r & 15);
    sW[i] = Wd[(size_t)gcol * KP + cc];
  }
#endif

  if (nt == 0) {  // zero both h parities (incl. K padding) and c
    for (int i = tid; i < 2 * B_ * KP; i += 256) h0[i] = __float2bfloat16(0.f);
    for (int i = tid; i < B_ * H_; i += 256) cd[i] = 0.f;
  }
  __syncthreads();

  unsigned phase = 0;
  half_grid_barrier(ctr, NBLK, ++phase);

  const int row0 = wave * 16;               // 8 waves x 16 rows = 128 batch rows
  const int col  = ht0 + (lane & 15);       // this lane's hidden column
  const int rb   = row0 + ((lane >> 4) * 8);

  for (int s = 0; s < T_; ++s) {
    const int t = dir ? (T_ - 1 - s) : s;
    const __hip_bfloat16* hr = (s & 1) ? h1 : h0;  // read h_{t-1}
    __hip_bfloat16* hw = (s & 1) ? h0 : h1;        // write h_t

    // -------- recurrent GEMM: c[g] = gates[t] + h_{t-1} @ Whh^T --------
    const float* Gt = Gd + (size_t)t * B_ * G4;
    f32x8 c[4];
#pragma unroll
    for (int g = 0; g < 4; ++g)
      c[g] = load_c_tile(Gt, G4, row0, g * H_ + ht0, lane);

#pragma unroll
    for (int k = 0; k < KP; k += 32) {
      bf16x16 a = load_a_frag(hr, KP, row0, k, lane);  // h (L2-resident, bf16)
#pragma unroll
      for (int g = 0; g < 4; ++g) {
        bf16x16 b = load_b_frag(sW + (size_t)g * 16 * KP, KP, 0, k, lane);
        c[g] = wmma_bf16(a, b, c[g]);
      }
    }

    // -------- pointwise LSTM cell, entirely from WMMA registers --------
    const int bt = bsz[t];
#pragma unroll
    for (int r = 0; r < 8; ++r) {
      const int row = rb + r;
      const float iv = fsig(c[0][r]);
      const float fv = fsig(c[1][r]);
      const float gv = tanhf(c[2][r]);
      const float ov = fsig(c[3][r]);
      const float cp = cd[row * H_ + col];
      const float cn = fv * cp + iv * gv;
      const float hn = ov * tanhf(cn);
      const float hp = __bfloat162float(hr[row * KP + col]);
      const bool  m  = row < bt;
      cd[row * H_ + col] = m ? cn : cp;
      hw[row * KP + col] = __float2bfloat16(m ? hn : hp);
      const float outv = m ? hn : 0.f;
      const size_t oi = ((size_t)t * B_ + row) * (2 * H_) + (size_t)dir * H_ + col;
      if (Xout) Xout[oi] = __float2bfloat16(outv);
      if (Fout) Fout[oi] = outv;
    }

    half_grid_barrier(ctr, NBLK, ++phase);
  }
}

// ---------------- host launch ----------------
static inline size_t rup256(size_t x) { return (x + 255) & ~(size_t)255; }

extern "C" void kernel_launch(void* const* d_in, const int* in_sizes, int n_in,
                              void* d_out, int out_size, void* d_ws, size_t ws_size,
                              hipStream_t stream) {
  (void)in_sizes; (void)n_in; (void)out_size; (void)ws_size;
  const float* x    = (const float*)d_in[0];   // [T,B,800]
  const float* Wih  = (const float*)d_in[1];   // [3,2,1600,800]
  const float* Whh  = (const float*)d_in[2];   // [3,2,1600,400]
  const float* bih  = (const float*)d_in[3];   // [3,2,1600]
  const float* bhh  = (const float*)d_in[4];   // [3,2,1600]
  const int*   bsz  = (const int*)d_in[5];     // [T]
  float* out = (float*)d_out;                  // [T,B,800]

  char* p = (char*)d_ws;
  auto take = [&](size_t bytes) { char* r = p; p += rup256(bytes); return (void*)r; };

  __hip_bfloat16* WihBF = (__hip_bfloat16*)take((size_t)L_ * 2 * G4 * IN_ * 2);
  __hip_bfloat16* WhhBF = (__hip_bfloat16*)take((size_t)L_ * 2 * G4 * KP * 2);
  float*          biasF = (float*)take((size_t)L_ * 2 * G4 * 4);
  __hip_bfloat16* XbfA  = (__hip_bfloat16*)take((size_t)MROWS * IN_ * 2);
  __hip_bfloat16* XbfB  = (__hip_bfloat16*)take((size_t)MROWS * IN_ * 2);
  float*          gates = (float*)take((size_t)2 * MROWS * G4 * 4);
  __hip_bfloat16* hping = (__hip_bfloat16*)take((size_t)2 * 2 * B_ * KP * 2);
  float*          cst   = (float*)take((size_t)2 * B_ * H_ * 4);
  unsigned*       ctrs  = (unsigned*)take((size_t)L_ * 2 * 4);

  zero_ctrs_kernel<<<1, 32, 0, stream>>>(ctrs, L_ * 2);
  cvt_f32_bf16_kernel<<<2048, 256, 0, stream>>>(x, XbfA, (long long)MROWS * IN_);
  cvt_f32_bf16_kernel<<<2048, 256, 0, stream>>>(Wih, WihBF, (long long)L_ * 2 * G4 * IN_);
  cvt_pad_whh_kernel<<<2048, 256, 0, stream>>>(Whh, WhhBF, L_ * 2 * G4);
  make_bias_kernel<<<(L_ * 2 * G4 + 255) / 256, 256, 0, stream>>>(bih, bhh, biasF, L_ * 2 * G4);

  __hip_bfloat16* Xcur = XbfA;
  __hip_bfloat16* Xnxt = XbfB;
  for (int l = 0; l < L_; ++l) {
    xproj_gemm_kernel<<<dim3(MROWS / 256, G4 / 64, 2), 256, 0, stream>>>(
        Xcur, WihBF + (size_t)l * 2 * G4 * IN_, biasF + (size_t)l * 2 * G4, gates);
    lstm_scan_kernel<<<2 * NBLK, 256, 0, stream>>>(
        gates, WhhBF + (size_t)l * 2 * G4 * KP, hping, cst,
        (l < L_ - 1) ? Xnxt : (__hip_bfloat16*)nullptr,
        (l == L_ - 1) ? out : (float*)nullptr,
        bsz, ctrs + l * 2);
    __hip_bfloat16* tmp = Xcur; Xcur = Xnxt; Xnxt = tmp;
  }
}
